// MixedContentHandler_26938034881235
// MI455X (gfx1250) — matrix operand order repaired and checked by reading, not compiled, not run on hardware
//
#include <hip/hip_runtime.h>
#include <hip/hip_bf16.h>
#include <math.h>

typedef __attribute__((ext_vector_type(16))) __bf16 v16bf;
typedef __attribute__((ext_vector_type(8)))  __bf16 v8bf;
typedef __attribute__((ext_vector_type(8)))  float  v8f;

#define NTOK   16384   // B*S
#define SEQ    2048
#define DMODEL 1024
#define NSEG   8       // segments of 256 along S for the ctx scan

// ---------------------------------------------------------------------------
// elementwise casts
// ---------------------------------------------------------------------------
__global__ __launch_bounds__(256) void copy_cast_kernel(const float* __restrict__ in,
                                                        float* __restrict__ out_f,
                                                        __bf16* __restrict__ out_b,
                                                        size_t n) {
  size_t i = (size_t)blockIdx.x * blockDim.x + threadIdx.x;
  size_t stride = (size_t)gridDim.x * blockDim.x;
  for (; i < n; i += stride) {
    float v = in[i];
    out_f[i] = v;
    out_b[i] = (__bf16)v;
  }
}

__global__ __launch_bounds__(256) void cast_kernel(const float* __restrict__ in,
                                                   __bf16* __restrict__ out_b,
                                                   size_t n) {
  size_t i = (size_t)blockIdx.x * blockDim.x + threadIdx.x;
  size_t stride = (size_t)gridDim.x * blockDim.x;
  for (; i < n; i += stride) out_b[i] = (__bf16)in[i];
}

// ---------------------------------------------------------------------------
// ctx: segmented scan over S.  partial[b][seg][d] then scan each segment.
// ---------------------------------------------------------------------------
__global__ __launch_bounds__(256) void ctx_partial_kernel(const float* __restrict__ x,
                                                          float* __restrict__ partial) {
  int bid = blockIdx.x;            // B * NSEG * (D/256) = 256 blocks
  int dc  = bid & 3;
  int seg = (bid >> 2) & 7;
  int b   = bid >> 5;
  int d   = dc * 256 + threadIdx.x;
  int s0  = seg * 256;
  float s = 0.f;
  for (int i = 0; i < 256; ++i)
    s += x[((size_t)b * SEQ + s0 + i) * DMODEL + d];
  partial[(b * NSEG + seg) * DMODEL + d] = s;
}

__global__ __launch_bounds__(256) void ctx_build_kernel(const float* __restrict__ x,
                                                        const float* __restrict__ partial,
                                                        __bf16* __restrict__ ctx_bf) {
  int bid = blockIdx.x;
  int dc  = bid & 3;
  int seg = (bid >> 2) & 7;
  int b   = bid >> 5;
  int d   = dc * 256 + threadIdx.x;

  float start = 0.f, total = 0.f;
  for (int j = 0; j < NSEG; ++j) {
    float p = partial[(b * NSEG + j) * DMODEL + d];
    total += p;
    if (j < seg) start += p;
  }
  float run = start;
  int s0 = seg * 256;
  for (int i = 0; i < 256; ++i) {
    int s = s0 + i;
    float xv = x[((size_t)b * SEQ + s) * DMODEL + d];
    float cs = run + xv;
    float pre = run;             // prefix sum (excl.)
    float suf = total - cs;      // suffix sum (excl.)
    float tf  = (float)s;
    float pm  = pre / fmaxf(tf, 1.f);
    float sm  = suf / fmaxf((float)(SEQ - 1) - tf, 1.f);
    float em  = (pre + suf) * (1.f / (float)(SEQ - 1));
    float ctx = (s == 0 || s == SEQ - 1) ? em : 0.5f * (pm + sm);
    ctx_bf[((size_t)b * SEQ + s) * DMODEL + d] = (__bf16)ctx;
    run = cs;
  }
}

// ---------------------------------------------------------------------------
// Generic bf16 WMMA GEMM:  out[M,N] = act( A[M,Ktot] @ W[Ktot,N] + bias )
// A split in two row-major segments A0 (K0 cols) / A1 (K1 cols).
// Block tile 128x128, 256 threads = 8 waves, wave = 16 rows x 128 cols.
// Double-buffered LDS (ping-pong), padded stride 40 bf16 vs bank conflicts,
// B stored transposed (Bs[n][k]) so fragments are contiguous b128 loads.
// ---------------------------------------------------------------------------
#define BM  128
#define BN  128
#define BK  32
#define BKP 40      // padded K stride in LDS

__global__ __launch_bounds__(256) void gemm_bf16_kernel(
    const __bf16* __restrict__ A0, const __bf16* __restrict__ A1,
    int K0, int K1,
    const __bf16* __restrict__ W, const float* __restrict__ bias,
    float* __restrict__ out, int N, int relu) {
  __shared__ __bf16 As[2][BM * BKP];   // 2 x 10 KB
  __shared__ __bf16 Bs[2][BN * BKP];   // 2 x 10 KB (transposed: [n][k])

  const int tid  = threadIdx.x;
  const int lane = tid & 31;
  const int wave = tid >> 5;
  const int half = lane >> 4;
  const int l16  = lane & 15;
  const int m0   = blockIdx.y * BM;
  const int n0   = blockIdx.x * BN;
  const int Ktot = K0 + K1;

  // per-thread staging coordinates (A: 2 chunks of 8 bf16; B: 2 chunks)
  const int ar0 = tid >> 2,            akc0 = (tid & 3) * 8;
  const int ar1 = (tid + 256) >> 2,    akc1 = ((tid + 256) & 3) * 8;
  const int bk0 = tid >> 4,            bnc0 = (tid & 15) * 8;
  const int bk1 = (tid + 256) >> 4,    bnc1 = ((tid + 256) & 15) * 8;

  v8f acc[8];
#pragma unroll
  for (int t = 0; t < 8; ++t)
#pragma unroll
    for (int r = 0; r < 8; ++r) acc[t][r] = 0.f;

  // ---- global load of one K-step's tiles into registers ----
  auto load_tiles = [&](int kb, uint4& a0, uint4& a1, uint4& b0, uint4& b1) {
    const __bf16* Abase;
    int lda, kloc;
    if (kb < K0) { Abase = A0; lda = K0; kloc = kb; }
    else         { Abase = A1; lda = K1; kloc = kb - K0; }
    a0 = *reinterpret_cast<const uint4*>(Abase + (size_t)(m0 + ar0) * lda + kloc + akc0);
    a1 = *reinterpret_cast<const uint4*>(Abase + (size_t)(m0 + ar1) * lda + kloc + akc1);
    b0 = *reinterpret_cast<const uint4*>(W + (size_t)(kb + bk0) * N + n0 + bnc0);
    b1 = *reinterpret_cast<const uint4*>(W + (size_t)(kb + bk1) * N + n0 + bnc1);
  };
  // ---- store staged registers into LDS buffer (B transposed) ----
  auto store_tiles = [&](int buf, const uint4& a0, const uint4& a1,
                         const uint4& b0, const uint4& b1) {
    *reinterpret_cast<uint4*>(&As[buf][ar0 * BKP + akc0]) = a0;
    *reinterpret_cast<uint4*>(&As[buf][ar1 * BKP + akc1]) = a1;
    const __bf16* w0 = reinterpret_cast<const __bf16*>(&b0);
    const __bf16* w1 = reinterpret_cast<const __bf16*>(&b1);
#pragma unroll
    for (int j = 0; j < 8; ++j) Bs[buf][(bnc0 + j) * BKP + bk0] = w0[j];
#pragma unroll
    for (int j = 0; j < 8; ++j) Bs[buf][(bnc1 + j) * BKP + bk1] = w1[j];
  };

  // prologue: stage K-step 0 into buffer 0
  {
    uint4 a0, a1, b0, b1;
    load_tiles(0, a0, a1, b0, b1);
    store_tiles(0, a0, a1, b0, b1);
  }
  __syncthreads();

  int buf = 0;
  for (int kb = 0; kb < Ktot; kb += BK, buf ^= 1) {
    const bool has_next = (kb + BK) < Ktot;
    uint4 na0, na1, nb0, nb1;
    if (has_next) load_tiles(kb + BK, na0, na1, nb0, nb1);   // in flight during compute
    if (kb + 2 * BK < Ktot) {                                 // warm L2 two steps ahead
      const __bf16* Ap; int lda2, kloc2, kb2 = kb + 2 * BK;
      if (kb2 < K0) { Ap = A0; lda2 = K0; kloc2 = kb2; }
      else          { Ap = A1; lda2 = K1; kloc2 = kb2 - K0; }
      __builtin_prefetch(Ap + (size_t)(m0 + ar0) * lda2 + kloc2 + akc0, 0, 0);
      __builtin_prefetch(W + (size_t)(kb2 + bk0) * N + n0 + bnc0, 0, 0);
    }

    // ---- compute from current buffer ----
    int arow = wave * 16 + l16;
    v8bf alo = *reinterpret_cast<const v8bf*>(&As[buf][arow * BKP + half * 16]);
    v8bf ahi = *reinterpret_cast<const v8bf*>(&As[buf][arow * BKP + half * 16 + 8]);
    v16bf afrag;
#pragma unroll
    for (int i = 0; i < 8; ++i) { afrag[i] = alo[i]; afrag[i + 8] = ahi[i]; }

#pragma unroll
    for (int t = 0; t < 8; ++t) {
      int n = t * 16 + l16;
      v8bf blo = *reinterpret_cast<const v8bf*>(&Bs[buf][n * BKP + half * 16]);
      v8bf bhi = *reinterpret_cast<const v8bf*>(&Bs[buf][n * BKP + half * 16 + 8]);
      v16bf bfrag;
#pragma unroll
      for (int i = 0; i < 8; ++i) { bfrag[i] = blo[i]; bfrag[i + 8] = bhi[i]; }
      acc[t] = __builtin_amdgcn_wmma_f32_16x16x32_bf16(
          false, afrag, false, bfrag, (short)0, acc[t], false, false);
    }

    // ---- store next K-step into the other buffer; single barrier per iter ----
    if (has_next) store_tiles(buf ^ 1, na0, na1, nb0, nb1);
    __syncthreads();
  }

  // ---- epilogue: C layout VGPR r -> rows r (lanes 0-15) / r+8 (lanes 16-31)
#pragma unroll
  for (int t = 0; t < 8; ++t) {
    int n = n0 + t * 16 + l16;
    float bv = bias ? bias[n] : 0.f;
#pragma unroll
    for (int r = 0; r < 8; ++r) {
      int m = m0 + wave * 16 + r + half * 8;
      float v = acc[t][r] + bv;
      if (relu) v = fmaxf(v, 0.f);
      out[(size_t)m * N + n] = v;
    }
  }
}

// ---------------------------------------------------------------------------
// wave-level reduction helper
// ---------------------------------------------------------------------------
__device__ __forceinline__ float wred(float v) {
  for (int o = 16; o > 0; o >>= 1) v += __shfl_down(v, o, 32);
  return v;
}
__device__ __forceinline__ float sigmoidf(float x) { return 1.f / (1.f + expf(-x)); }

// ---------------------------------------------------------------------------
// content_types = softmax(h1 @ W_cc2 + b_cc2); relevance = 1 - ct[3]
// one wave per token, K = 512, N = 5
// ---------------------------------------------------------------------------
__global__ __launch_bounds__(256) void content_types_kernel(
    const float* __restrict__ h1, const float* __restrict__ Wcc2,
    const float* __restrict__ bcc2, float* __restrict__ out_ct,
    float* __restrict__ out_rel) {
  __shared__ float w[512 * 5];
  __shared__ float bb[5];
  int tid = threadIdx.x;
  for (int i = tid; i < 512 * 5; i += 256) w[i] = Wcc2[i];
  if (tid < 5) bb[tid] = bcc2[tid];
  __syncthreads();

  int wave = tid >> 5, lane = tid & 31;
  int t = blockIdx.x * 8 + wave;
  float acc[5] = {0.f, 0.f, 0.f, 0.f, 0.f};
  for (int k = lane; k < 512; k += 32) {
    float v = h1[(size_t)t * 512 + k];
#pragma unroll
    for (int c = 0; c < 5; ++c) acc[c] += v * w[k * 5 + c];
  }
#pragma unroll
  for (int c = 0; c < 5; ++c) acc[c] = wred(acc[c]);
  if (lane == 0) {
    float l[5], mx = -1e30f;
#pragma unroll
    for (int c = 0; c < 5; ++c) { l[c] = acc[c] + bb[c]; mx = fmaxf(mx, l[c]); }
    float s = 0.f;
#pragma unroll
    for (int c = 0; c < 5; ++c) { l[c] = expf(l[c] - mx); s += l[c]; }
    float inv = 1.f / s;
#pragma unroll
    for (int c = 0; c < 5; ++c) out_ct[(size_t)t * 5 + c] = l[c] * inv;
    out_rel[t] = 1.f - l[3] * inv;
  }
}

// ---------------------------------------------------------------------------
// in-place LayerNorm + exact GELU over D=1024, one block per token
// ---------------------------------------------------------------------------
__global__ __launch_bounds__(256) void ln_gelu_kernel(float* __restrict__ h,
                                                      const float* __restrict__ g,
                                                      const float* __restrict__ be) {
  __shared__ float red[256];
  float* hp = h + (size_t)blockIdx.x * DMODEL;
  int tid = threadIdx.x;
  float v[4];
  float s = 0.f;
#pragma unroll
  for (int j = 0; j < 4; ++j) { v[j] = hp[tid + j * 256]; s += v[j]; }
  red[tid] = s; __syncthreads();
  for (int off = 128; off > 0; off >>= 1) {
    if (tid < off) red[tid] += red[tid + off];
    __syncthreads();
  }
  float mu = red[0] * (1.f / (float)DMODEL);
  __syncthreads();
  float q = 0.f;
#pragma unroll
  for (int j = 0; j < 4; ++j) { float d = v[j] - mu; q += d * d; }
  red[tid] = q; __syncthreads();
  for (int off = 128; off > 0; off >>= 1) {
    if (tid < off) red[tid] += red[tid + off];
    __syncthreads();
  }
  float var = red[0] * (1.f / (float)DMODEL);
  float rs = rsqrtf(var + 1e-5f);
#pragma unroll
  for (int j = 0; j < 4; ++j) {
    int idx = tid + j * 256;
    float hv = (v[j] - mu) * rs * g[idx] + be[idx];
    hp[idx] = 0.5f * hv * (1.f + erff(hv * 0.70710678118654752f));  // exact gelu
  }
}

// ---------------------------------------------------------------------------
// enhanced = max(sigmoid(h_cd_gelu @ W_cd2 + b), ct[:,2]); wave/token, K=1024
// ---------------------------------------------------------------------------
__global__ __launch_bounds__(256) void contr_enh_kernel(
    const float* __restrict__ hcd, const float* __restrict__ Wcd2,
    const float* __restrict__ bcd2, const float* __restrict__ ct,
    float* __restrict__ out_enh) {
  int wave = threadIdx.x >> 5, lane = threadIdx.x & 31;
  int t = blockIdx.x * 8 + wave;
  float acc = 0.f;
  for (int k = lane; k < DMODEL; k += 32)
    acc += hcd[(size_t)t * DMODEL + k] * Wcd2[k];
  acc = wred(acc);
  if (lane == 0) {
    float c = sigmoidf(acc + bcd2[0]);
    out_enh[t] = fmaxf(c, ct[(size_t)t * 5 + 2]);
  }
}

// ---------------------------------------------------------------------------
// weighted_bias = sigmoid(h_be @ W_be2 + b) * (ct @ bias_weights); K=256
// ---------------------------------------------------------------------------
__global__ __launch_bounds__(256) void wbias_kernel(
    const float* __restrict__ hbe, const float* __restrict__ Wbe2,
    const float* __restrict__ bbe2, const float* __restrict__ ct,
    float* __restrict__ out_wb) {
  int wave = threadIdx.x >> 5, lane = threadIdx.x & 31;
  int t = blockIdx.x * 8 + wave;
  float acc = 0.f;
  for (int k = lane; k < 256; k += 32)
    acc += hbe[(size_t)t * 256 + k] * Wbe2[k];
  acc = wred(acc);
  if (lane == 0) {
    float bs = sigmoidf(acc + bbe2[0]);
    const float* c = &ct[(size_t)t * 5];
    float wsum = 0.1f * c[0] + 1.0f * c[1] + 0.8f * c[2] + 0.3f * c[3] + 0.5f * c[4];
    out_wb[t] = bs * wsum;
  }
}

// ---------------------------------------------------------------------------
// h_cr += ct @ W_cr1[1024:1029,:]  then ReLU (in place); elementwise
// ---------------------------------------------------------------------------
__global__ __launch_bounds__(256) void cr_tail_kernel(float* __restrict__ hcr,
                                                      const float* __restrict__ ct,
                                                      const float* __restrict__ Wcr1) {
  size_t idx = (size_t)blockIdx.x * 256 + threadIdx.x;  // NTOK*512 total
  int t = (int)(idx >> 9);
  int n = (int)(idx & 511);
  float v = hcr[idx];
  const float* c = &ct[(size_t)t * 5];
#pragma unroll
  for (int cc = 0; cc < 5; ++cc)
    v += c[cc] * Wcr1[(size_t)(1024 + cc) * 512 + n];
  hcr[idx] = fmaxf(v, 0.f);
}

// ---------------------------------------------------------------------------
// credibility = sigmoid(h_cr @ W_cr2 + b); wave/token, K=512
// ---------------------------------------------------------------------------
__global__ __launch_bounds__(256) void cred_kernel(
    const float* __restrict__ hcr, const float* __restrict__ Wcr2,
    const float* __restrict__ bcr2, float* __restrict__ out_cred) {
  int wave = threadIdx.x >> 5, lane = threadIdx.x & 31;
  int t = blockIdx.x * 8 + wave;
  float acc = 0.f;
  for (int k = lane; k < 512; k += 32)
    acc += hcr[(size_t)t * 512 + k] * Wcr2[k];
  acc = wred(acc);
  if (lane == 0) out_cred[t] = sigmoidf(acc + bcr2[0]);
}

// ---------------------------------------------------------------------------
// host launcher
// ---------------------------------------------------------------------------
extern "C" void kernel_launch(void* const* d_in, const int* in_sizes, int n_in,
                              void* d_out, int out_size, void* d_ws, size_t ws_size,
                              hipStream_t stream) {
  (void)in_sizes; (void)n_in; (void)out_size; (void)ws_size;
  const float* x     = (const float*)d_in[0];
  const float* W_cc1 = (const float*)d_in[1];
  const float* b_cc1 = (const float*)d_in[2];
  const float* W_cc2 = (const float*)d_in[3];
  const float* b_cc2 = (const float*)d_in[4];
  const float* W_cd1 = (const float*)d_in[5];
  const float* b_cd1 = (const float*)d_in[6];
  const float* g_ln  = (const float*)d_in[7];
  const float* be_ln = (const float*)d_in[8];
  const float* W_cd2 = (const float*)d_in[9];
  const float* b_cd2 = (const float*)d_in[10];
  const float* W_be1 = (const float*)d_in[11];
  const float* b_be1 = (const float*)d_in[12];
  const float* W_be2 = (const float*)d_in[13];
  const float* b_be2 = (const float*)d_in[14];
  const float* W_cr1 = (const float*)d_in[15];
  const float* b_cr1 = (const float*)d_in[16];
  const float* W_cr2 = (const float*)d_in[17];
  const float* b_cr2 = (const float*)d_in[18];

  float* out = (float*)d_out;
  // output layout (floats)
  float* out_x    = out;                       // 16,777,216
  float* out_ct   = out + 16777216;            //     81,920
  float* out_enh  = out + 16859136;            //     16,384
  float* out_wb   = out + 16875520;            //     16,384
  float* out_rel  = out + 16891904;            //     16,384
  float* out_cred = out + 16908288;            //     16,384

  // workspace layout (bytes)
  char* ws = (char*)d_ws;
  __bf16* x_bf    = (__bf16*)(ws + 0);                   // 32 MB
  __bf16* ctx_bf  = (__bf16*)(ws + 33554432);            // 32 MB
  __bf16* Wcc1_bf = (__bf16*)(ws + 67108864);            //  1 MB
  __bf16* Wcd1_bf = (__bf16*)(ws + 68157440);            //  4 MB
  __bf16* Wbe1_bf = (__bf16*)(ws + 72351744);            // .5 MB
  __bf16* Wcr1_bf = (__bf16*)(ws + 72876032);            //  1 MB (rows 0..1023)
  float*  partial = (float*)(ws + 73924608);             // 256 KB
  float*  h1      = (float*)(ws + 74186752);             // 32 MB
  float*  h_be    = (float*)(ws + 107741184);            // 16 MB
  float*  h_cd    = (float*)(ws + 124518400);            // 64 MB
  float*  h_cr    = (float*)(ws + 191627264);            // 32 MB

  // 1) copy x -> out, cast x -> bf16
  copy_cast_kernel<<<8192, 256, 0, stream>>>(x, out_x, x_bf, (size_t)NTOK * DMODEL);

  // 2) cast weight matrices to bf16
  cast_kernel<<<2048, 256, 0, stream>>>(W_cc1, Wcc1_bf, (size_t)1024 * 512);
  cast_kernel<<<2048, 256, 0, stream>>>(W_cd1, Wcd1_bf, (size_t)2048 * 1024);
  cast_kernel<<<1024, 256, 0, stream>>>(W_be1, Wbe1_bf, (size_t)1024 * 256);
  cast_kernel<<<2048, 256, 0, stream>>>(W_cr1, Wcr1_bf, (size_t)1024 * 512); // first 1024 rows

  // 3) ctx scan -> ctx_bf
  ctx_partial_kernel<<<256, 256, 0, stream>>>(x, partial);
  ctx_build_kernel<<<256, 256, 0, stream>>>(x, partial, ctx_bf);

  // 4) WMMA GEMMs (M = 16384, tile 128x128, double-buffered)
  gemm_bf16_kernel<<<dim3(512 / BN, NTOK / BM), 256, 0, stream>>>(
      x_bf, nullptr, 1024, 0, Wcc1_bf, b_cc1, h1, 512, 1);
  gemm_bf16_kernel<<<dim3(256 / BN, NTOK / BM), 256, 0, stream>>>(
      x_bf, nullptr, 1024, 0, Wbe1_bf, b_be1, h_be, 256, 1);
  gemm_bf16_kernel<<<dim3(1024 / BN, NTOK / BM), 256, 0, stream>>>(
      x_bf, ctx_bf, 1024, 1024, Wcd1_bf, b_cd1, h_cd, 1024, 0);
  gemm_bf16_kernel<<<dim3(512 / BN, NTOK / BM), 256, 0, stream>>>(
      x_bf, nullptr, 1024, 0, Wcr1_bf, b_cr1, h_cr, 512, 0);

  // 5) content types + relevance
  content_types_kernel<<<NTOK / 8, 256, 0, stream>>>(h1, W_cc2, b_cc2, out_ct, out_rel);

  // 6) LayerNorm + GELU in place on h_cd
  ln_gelu_kernel<<<NTOK, 256, 0, stream>>>(h_cd, g_ln, be_ln);

  // 7) contradiction -> enhanced
  contr_enh_kernel<<<NTOK / 8, 256, 0, stream>>>(h_cd, W_cd2, b_cd2, out_ct, out_enh);

  // 8) weighted bias
  wbias_kernel<<<NTOK / 8, 256, 0, stream>>>(h_be, W_be2, b_be2, out_ct, out_wb);

  // 9) credibility: ct-tail + relu, then head
  cr_tail_kernel<<<(NTOK * 512) / 256, 256, 0, stream>>>(h_cr, out_ct, W_cr1);
  cred_kernel<<<NTOK / 8, 256, 0, stream>>>(h_cr, W_cr2, b_cr2, out_cred);
}
